// GATLayer_72060961292563
// MI455X (gfx1250) — matrix operand order repaired and checked by reading, not compiled, not run on hardware
//
#include <hip/hip_runtime.h>
#include <hip/hip_bf16.h>

// ---- problem constants ----------------------------------------------------
#define N_NODES   100000
#define F_IN      128
#define K_NEIGH   16
#define N_HEADS   8
#define HEAD_D    16
#define N_TILES   (N_NODES / 16)     // 6250, exact
#define NF        ((size_t)N_NODES * F_IN)

typedef __bf16 bf16_t;
typedef __attribute__((ext_vector_type(16))) bf16_t v16bf;
typedef __attribute__((ext_vector_type(8)))  float  v8f;

// Packed-weight layout (per weight matrix, 128x128):
//   32 fragments (o=0..7, kc=0..3), each 32 lanes x 16 bf16, hi plane then lo
//   plane:  hi at frag*512 + lane*16, lo at +16384.  One weight = 32768 bf16.
#define PW_PER_W  32768
#define PW_LO     16384

// ---- bf16 split helpers ---------------------------------------------------
// A fragment: 16x32 (MxK) bf16, row-major fp32 source, leading dim ld.
__device__ inline void load_a_frag(const float* __restrict__ src, int ld,
                                   int row0, int k0, int lane,
                                   v16bf& ahi, v16bf& alo) {
    const int hf = lane >> 4;
    const int m  = lane & 15;
    const float* p = src + (size_t)(row0 + m) * ld;
#pragma unroll
    for (int j = 0; j < 16; ++j) {
        int k = k0 + hf * 16 + j;
        float a = p[k];
        bf16_t hi = (bf16_t)a;
        ahi[j] = hi;
        alo[j] = (bf16_t)(a - (float)hi);
    }
}

// B fragment from raw fp32 weight (used only by the prepack kernel).
// B[k][n] = W[n][k]; lane holds column n = lane&15, 16 contiguous K.
__device__ inline void load_b_frag(const float* __restrict__ W, int ld,
                                   int n0, int k0, int lane,
                                   v16bf& bhi, v16bf& blo) {
    const int hf = lane >> 4;
    const int n  = lane & 15;
    const float* p = W + (size_t)(n0 + n) * ld + k0 + hf * 16;
#pragma unroll
    for (int j = 0; j < 16; ++j) {
        float b = p[j];
        bf16_t hi = (bf16_t)b;
        bhi[j] = hi;
        blo[j] = (bf16_t)(b - (float)hi);
    }
}

// B fragment from prepacked buffer: one aligned 32B vector load per plane.
__device__ inline void load_b_packed(const bf16_t* __restrict__ base,
                                     int o, int kc, int lane,
                                     v16bf& bhi, v16bf& blo) {
    const bf16_t* hp = base + ((o * 4 + kc) * 512 + lane * 16);
    bhi = *(const v16bf*)hp;
    blo = *(const v16bf*)(hp + PW_LO);
}

// split-bf16 near-fp32 product: a*b ~= ah*bh + al*bh + ah*bl (f32 accum)
__device__ inline v8f wmma3(const v16bf& ah, const v16bf& al,
                            const v16bf& bh, const v16bf& bl, v8f c) {
    c = __builtin_amdgcn_wmma_f32_16x16x32_bf16(false, ah, false, bh, (short)0, c, false, false);
    c = __builtin_amdgcn_wmma_f32_16x16x32_bf16(false, al, false, bh, (short)0, c, false, false);
    c = __builtin_amdgcn_wmma_f32_16x16x32_bf16(false, ah, false, bl, (short)0, c, false, false);
    return c;
}

// ---- kernel 0: prepack Wq/Wk/Wv/Wo into bf16 hi/lo fragment layout --------
// grid = 4 weights * 32 fragments = 128 waves of 32.
__global__ __launch_bounds__(32) void prepack_kernel(
        const float* __restrict__ Wq, const float* __restrict__ Wk,
        const float* __restrict__ Wv, const float* __restrict__ Wo,
        bf16_t* __restrict__ pack) {
    const int f    = blockIdx.x;
    const int w    = f >> 5;           // 0..3
    const int idx  = f & 31;           // o*4 + kc
    const int o    = idx >> 2;
    const int kc   = idx & 3;
    const int lane = threadIdx.x & 31;
    const float* W = (w == 0) ? Wq : (w == 1) ? Wk : (w == 2) ? Wv : Wo;

    v16bf bhi, blo;
    load_b_frag(W, F_IN, o * 16, kc * 32, lane, bhi, blo);

    bf16_t* hp = pack + (size_t)w * PW_PER_W + (size_t)idx * 512 + lane * 16;
    *(v16bf*)hp            = bhi;
    *(v16bf*)(hp + PW_LO)  = blo;
}

// ---- kernel 1: Q/K/V = h @ W{q,k,v}.T  ------------------------------------
// one wave per 16-row tile; A of h loaded+split ONCE, reused for all 3
// weights. qkv buffer holds Q | K | V contiguously (out = qkv + w*NF), so
// all stores stay in provably-global address space.
__global__ __launch_bounds__(32) void qkv_kernel(
        const float* __restrict__ h, const bf16_t* __restrict__ pack,
        float* __restrict__ qkv) {
    const int row0 = blockIdx.x * 16;
    const int lane = threadIdx.x & 31;
    const int hf   = lane >> 4;
    const int nn   = lane & 15;

    v16bf ahi[4], alo[4];
#pragma unroll
    for (int kc = 0; kc < 4; ++kc)
        load_a_frag(h, F_IN, row0, kc * 32, lane, ahi[kc], alo[kc]);

#pragma unroll 1
    for (int w = 0; w < 3; ++w) {
        const bf16_t* base = pack + (size_t)w * PW_PER_W;
        float* out = qkv + (size_t)w * NF + (size_t)row0 * F_IN;
#pragma unroll
        for (int o = 0; o < 8; ++o) {
            v8f c = {};
#pragma unroll
            for (int kc = 0; kc < 4; ++kc) {
                v16bf bhi, blo;
                load_b_packed(base, o, kc, lane, bhi, blo);
                c = wmma3(ahi[kc], alo[kc], bhi, blo, c);
            }
            float* op = out + o * 16 + nn;
#pragma unroll
            for (int r = 0; r < 8; ++r)
                op[(size_t)(hf * 8 + r) * F_IN] = c[r];
        }
    }
}

// ---- kernel 2: neighbor attention (fp32 VALU) -----------------------------
// 128 threads / 16 nodes; thread = (node, head). Overwrites Q with attn-out.
__global__ __launch_bounds__(128) void attn_kernel(
        const int* __restrict__ nbrs,
        float* __restrict__ Q,               // in: Q, out: attn output
        const float* __restrict__ Kb, const float* __restrict__ Vb) {
    const int t    = threadIdx.x;
    const int node = blockIdx.x * 16 + (t >> 3);
    const int head = t & 7;

    const float* qp = Q + (size_t)node * F_IN + head * HEAD_D;
    float q[HEAD_D];
#pragma unroll
    for (int d = 0; d < HEAD_D; ++d) q[d] = qp[d];

    int nb[K_NEIGH];
#pragma unroll
    for (int k = 0; k < K_NEIGH; ++k) nb[k] = nbrs[node * K_NEIGH + k];

    float logit[K_NEIGH];
    float mx = -3.0e38f;
#pragma unroll
    for (int k = 0; k < K_NEIGH; ++k) {
        const float* kp = Kb + (size_t)nb[k] * F_IN + head * HEAD_D;
        float s = 0.f;
#pragma unroll
        for (int d = 0; d < HEAD_D; ++d) s = fmaf(q[d], kp[d], s);
        s *= 0.25f;                           // D^-0.5 = 1/sqrt(16)
        logit[k] = s;
        mx = fmaxf(mx, s);
    }
    float sum = 0.f;
#pragma unroll
    for (int k = 0; k < K_NEIGH; ++k) {
        logit[k] = __expf(logit[k] - mx);
        sum += logit[k];
    }
    const float inv = 1.f / sum;

    float acc[HEAD_D];
#pragma unroll
    for (int d = 0; d < HEAD_D; ++d) acc[d] = 0.f;
#pragma unroll
    for (int k = 0; k < K_NEIGH; ++k) {
        const float* vp = Vb + (size_t)nb[k] * F_IN + head * HEAD_D;
        const float a = logit[k] * inv;
#pragma unroll
        for (int d = 0; d < HEAD_D; ++d) acc[d] = fmaf(a, vp[d], acc[d]);
    }
    float* op = Q + (size_t)node * F_IN + head * HEAD_D;
#pragma unroll
    for (int d = 0; d < HEAD_D; ++d) op[d] = acc[d];
}

// ---- kernel 3: out-proj GEMM + bias + residual + LayerNorm ----------------
// 4 waves per 16-row tile: each wave does 2 of 8 column tiles via WMMA.
__global__ __launch_bounds__(128) void oproj_ln_kernel(
        const float* __restrict__ x,          // attn output [N,128]
        const float* __restrict__ h, const bf16_t* __restrict__ packWo,
        const float* __restrict__ bo,
        const float* __restrict__ gamma, const float* __restrict__ beta,
        float* __restrict__ out) {
    __shared__ float y[16][F_IN];
    __shared__ float s_mu[16], s_rstd[16];

    const int row0 = blockIdx.x * 16;
    const int t    = threadIdx.x;
    const int wave = t >> 5;
    const int lane = t & 31;
    const int hf   = lane >> 4;
    const int nn   = lane & 15;

    v16bf ahi[4], alo[4];
#pragma unroll
    for (int kc = 0; kc < 4; ++kc)
        load_a_frag(x, F_IN, row0, kc * 32, lane, ahi[kc], alo[kc]);

#pragma unroll
    for (int oo = 0; oo < 2; ++oo) {
        const int o = wave + oo * 4;
        v8f c = {};
#pragma unroll
        for (int kc = 0; kc < 4; ++kc) {
            v16bf bhi, blo;
            load_b_packed(packWo, o, kc, lane, bhi, blo);
            c = wmma3(ahi[kc], alo[kc], bhi, blo, c);
        }
        const int col = o * 16 + nn;
        const float bias = bo[col];
#pragma unroll
        for (int r = 0; r < 8; ++r) {
            const int row = hf * 8 + r;
            y[row][col] = c[r] + bias + h[(size_t)(row0 + row) * F_IN + col];
        }
    }
    __syncthreads();

    if (t < 16) {
        float s = 0.f, s2 = 0.f;
#pragma unroll 8
        for (int cidx = 0; cidx < F_IN; ++cidx) {
            float v = y[t][cidx];
            s += v; s2 += v * v;
        }
        const float mu  = s * (1.f / F_IN);
        const float var = s2 * (1.f / F_IN) - mu * mu;
        s_mu[t]   = mu;
        s_rstd[t] = rsqrtf(var + 1e-5f);
    }
    __syncthreads();

    const int row = t >> 3;
    const int c0  = (t & 7) * 16;
    const float mu = s_mu[row], rstd = s_rstd[row];
    float* op = out + (size_t)(row0 + row) * F_IN;
#pragma unroll
    for (int j = 0; j < 16; ++j) {
        const int cidx = c0 + j;
        op[cidx] = (y[row][cidx] - mu) * rstd * gamma[cidx] + beta[cidx];
    }
}

// ---- launcher -------------------------------------------------------------
extern "C" void kernel_launch(void* const* d_in, const int* in_sizes, int n_in,
                              void* d_out, int out_size, void* d_ws, size_t ws_size,
                              hipStream_t stream) {
    const float* h     = (const float*)d_in[0];
    const int*   nbrs  = (const int*)  d_in[1];
    const float* Wq    = (const float*)d_in[2];
    const float* Wk    = (const float*)d_in[3];
    const float* Wv    = (const float*)d_in[4];
    const float* Wo    = (const float*)d_in[5];
    const float* bo    = (const float*)d_in[6];
    const float* gamma = (const float*)d_in[7];
    const float* beta  = (const float*)d_in[8];
    float* out = (float*)d_out;

    // workspace: qkv = Q|K|V contiguous, 3*N*128 f32 = 153.6 MB,
    // then packed weights: 4 * 32768 bf16 = 256 KB (32B-aligned by layout).
    float* qkv = (float*)d_ws;
    float* Q   = qkv;
    float* Kb  = qkv + NF;
    float* Vb  = qkv + 2 * NF;
    bf16_t* pack = (bf16_t*)(qkv + 3 * NF);

    prepack_kernel<<<128, 32, 0, stream>>>(Wq, Wk, Wv, Wo, pack);
    qkv_kernel<<<N_TILES, 32, 0, stream>>>(h, pack, qkv);
    attn_kernel<<<N_TILES, 128, 0, stream>>>(nbrs, Q, Kb, Vb);
    oproj_ln_kernel<<<N_TILES, 128, 0, stream>>>(
        Q, h, pack + (size_t)3 * PW_PER_W, bo, gamma, beta, out);
}
